// Attend_15942918603084
// MI455X (gfx1250) — matrix operand order repaired and checked by reading, not compiled, not run on hardware
//
#include <hip/hip_runtime.h>
#include <float.h>

typedef __attribute__((ext_vector_type(16))) _Float16 v16h;
typedef __attribute__((ext_vector_type(8)))  float    v8f;

#define BDIM 2
#define HN   16
#define SN   2048
#define DN   64
#define LSTRIDE 2052          // 2048 + 4 pad floats: row stride not a multiple of 64 banks
#define CAP    50.0f
// folded: (1/sqrt(64)) * (1/SOFTCLAMP) = 0.125 * 0.02
#define SCALE_OVER_CAP 0.0025f
#define LOG2E  1.4426950408889634f

// Native CDNA5 tanh (VOP1 TRANS op). v_nop appended to respect the TRANS
// "1 independent op before result consumption" hazard, since the compiler
// cannot classify inline asm as a TRANS op.
__device__ __forceinline__ float tanh_hw(float x) {
    float y;
    asm("v_tanh_f32 %0, %1\n\tv_nop" : "=v"(y) : "v"(x));
    return y;
}

__global__ __launch_bounds__(256) void attend_kernel(
    const float* __restrict__ q, const float* __restrict__ k,
    const float* __restrict__ v, const unsigned char* __restrict__ mask,
    float* __restrict__ out, float* __restrict__ attn)
{
    __shared__ float logits[16 * LSTRIDE];   // 128.25 KB: full 16 x 2048 logit rows
    __shared__ float qs[16 * DN];            // staged Q tile
    __shared__ float red[16 * 16];           // row reduction scratch
    __shared__ float rowstat[16];            // row max, then row sum

    const int tid  = threadIdx.x;
    const int lane = tid & 31;
    const int wv   = tid >> 5;     // wave 0..7
    const int lh   = lane & 15;    // lane % 16
    const int hi   = lane >> 4;    // lane half (0/1)

    const int wg = blockIdx.x;
    const int qb = wg & (SN / 16 - 1);   // 128 query blocks
    const int bh = wg >> 7;              // b*H + h  (0..31)
    const int b  = bh >> 4;
    const int q0 = qb * 16;

    const size_t bh_off = (size_t)bh * SN * DN;
    const float* qp = q + bh_off + (size_t)q0 * DN;
    const float* kp = k + bh_off;
    const float* vp = v + bh_off;
    const unsigned char* mp = mask + b * SN;

    // ---- stage Q tile (16x64 fp32) into LDS, coalesced ----
    for (int i = tid; i < 16 * DN; i += 256) qs[i] = qp[i];
    __syncthreads();

    // ---- build f16 A-fragments for Q (16x32, contiguous k = 16*hi + s, same order as B) ----
    const int koff = hi * 16;
    v16h a0, a1;
#pragma unroll
    for (int j = 0; j < 8; ++j) {
        a0[j]     = (_Float16)qs[lh * DN + koff + j];          // K = koff+j
        a0[j + 8] = (_Float16)qs[lh * DN + 8 + koff + j];      // K = 8+koff+j
        a1[j]     = (_Float16)qs[lh * DN + 32 + koff + j];
        a1[j + 8] = (_Float16)qs[lh * DN + 40 + koff + j];
    }

    // ---- Phase A: logits = softcap(scale * Q K^T), masked; each wave owns 256 keys ----
    for (int t = 0; t < 16; ++t) {
        const int key0 = (wv * 16 + t) * 16;
        const float* krow = kp + (size_t)(key0 + lh) * DN;     // B[kk][n] = K[key0+n][kk]
        const int cb = hi * 16;
        v16h b0, b1;
#pragma unroll
        for (int j = 0; j < 16; ++j) {
            b0[j] = (_Float16)krow[cb + j];                    // d = cb+j        (K-chunk 0..31)
            b1[j] = (_Float16)krow[32 + cb + j];               // d = 32+cb+j     (K-chunk 32..63)
        }
        v8f acc = {};
        acc = __builtin_amdgcn_wmma_f32_16x16x32_f16(false, a0, false, b0, (short)0, acc, false, false);
        acc = __builtin_amdgcn_wmma_f32_16x16x32_f16(false, a1, false, b1, (short)0, acc, false, false);

        const bool keep = mp[key0 + lh] != 0;                  // column N = lh
#pragma unroll
        for (int r = 0; r < 8; ++r) {
            // tanh((raw/8)/50)*50 : one fused constant mul + native v_tanh + one mul
            float s = tanh_hw(acc[r] * SCALE_OVER_CAP) * CAP;
            s = keep ? s : -FLT_MAX;
            logits[(r + hi * 8) * LSTRIDE + key0 + lh] = s;    // row M = r + 8*hi
        }
    }
    __syncthreads();

    // ---- Phase B: row-wise softmax over 2048 cols (16 threads per row) ----
    const int row = tid >> 4;
    const int idx = tid & 15;

    float m = -FLT_MAX;
    for (int c = idx; c < SN; c += 16) m = fmaxf(m, logits[row * LSTRIDE + c]);
    red[row * 16 + idx] = m;
    __syncthreads();
    if (idx == 0) {
        float mm = red[row * 16];
#pragma unroll
        for (int i = 1; i < 16; ++i) mm = fmaxf(mm, red[row * 16 + i]);
        rowstat[row] = mm;
    }
    __syncthreads();
    const float rm = rowstat[row];

    float sum = 0.f;
    for (int c = idx; c < SN; c += 16) {
        // arg is always <= 0: single v_exp_f32, no range fixup needed.
        // masked: (-FLT_MAX - rm)*log2e -> -huge -> exp2 = 0
        // all-masked row: (rm - rm) = 0 -> exp2 = 1 (uniform, matches reference)
        float p = __builtin_amdgcn_exp2f((logits[row * LSTRIDE + c] - rm) * LOG2E);
        logits[row * LSTRIDE + c] = p;
        sum += p;
    }
    red[row * 16 + idx] = sum;
    __syncthreads();
    if (idx == 0) {
        float ss = 0.f;
#pragma unroll
        for (int i = 0; i < 16; ++i) ss += red[row * 16 + i];
        rowstat[row] = ss;
    }
    __syncthreads();
    const float invl = __builtin_amdgcn_rcpf(rowstat[row]);

    // normalize; stream attn to HBM non-temporally (512 MB, write-once) and keep P in LDS
    float* attn_row = attn + ((size_t)(bh * SN + q0 + row)) * SN;
    for (int c = idx; c < SN; c += 16) {
        float p = logits[row * LSTRIDE + c] * invl;
        logits[row * LSTRIDE + c] = p;
        __builtin_nontemporal_store(p, attn_row + c);
    }
    __syncthreads();

    // ---- Phase D: O = P V ; each wave accumulates its 256 keys, 4 D-tiles ----
    v8f c0 = {}, c1 = {}, c2 = {}, c3 = {};
    for (int step = 0; step < 8; ++step) {
        const int k0 = wv * 256 + step * 32;
        v16h pa;
#pragma unroll
        for (int j = 0; j < 8; ++j) {
            pa[j]     = (_Float16)logits[lh * LSTRIDE + k0 + koff + j];
            pa[j + 8] = (_Float16)logits[lh * LSTRIDE + k0 + 8 + koff + j];
        }
        const float* vb = vp + (size_t)(k0 + hi * 16) * DN + lh;   // B[kk][n] = V[k0+kk][d0+n]
        v16h b0, b1, b2, b3;
#pragma unroll
        for (int j = 0; j < 16; ++j) {
            const float* vr = vb + (size_t)j * DN;
            b0[j] = (_Float16)vr[0];
            b1[j] = (_Float16)vr[16];
            b2[j] = (_Float16)vr[32];
            b3[j] = (_Float16)vr[48];
        }
        c0 = __builtin_amdgcn_wmma_f32_16x16x32_f16(false, pa, false, b0, (short)0, c0, false, false);
        c1 = __builtin_amdgcn_wmma_f32_16x16x32_f16(false, pa, false, b1, (short)0, c1, false, false);
        c2 = __builtin_amdgcn_wmma_f32_16x16x32_f16(false, pa, false, b2, (short)0, c2, false, false);
        c3 = __builtin_amdgcn_wmma_f32_16x16x32_f16(false, pa, false, b3, (short)0, c3, false, false);
    }
    __syncthreads();                 // all waves done reading P from logits

    // cross-wave reduction through LDS (reuse logits area: 8*16*64 floats = 32 KB)
    float* ored = logits;
#pragma unroll
    for (int r = 0; r < 8; ++r) {
        const int M = r + hi * 8;
        float* dst = ored + (wv * 16 + M) * 64 + lh;
        dst[0]  = c0[r];
        dst[16] = c1[r];
        dst[32] = c2[r];
        dst[48] = c3[r];
    }
    __syncthreads();

    float* outp = out + bh_off + (size_t)q0 * DN;
    for (int o = tid; o < 16 * 64; o += 256) {
        float acc = 0.f;
#pragma unroll
        for (int w = 0; w < 8; ++w) acc += ored[(w * 16 + (o >> 6)) * 64 + (o & 63)];
        __builtin_nontemporal_store(acc, outp + o);   // coalesced fp32 stream-out
    }
}

extern "C" void kernel_launch(void* const* d_in, const int* in_sizes, int n_in,
                              void* d_out, int out_size, void* d_ws, size_t ws_size,
                              hipStream_t stream) {
    const float* q = (const float*)d_in[0];
    const float* k = (const float*)d_in[1];
    const float* v = (const float*)d_in[2];
    const unsigned char* mask = (const unsigned char*)d_in[3];   // jnp.bool_ = 1 byte/elem

    float* out  = (float*)d_out;
    float* attn = out + (size_t)BDIM * HN * SN * DN;             // tuple order: (out, attn)

    dim3 grid(BDIM * HN * (SN / 16));   // 4096 workgroups
    attend_kernel<<<grid, 256, 0, stream>>>(q, k, v, mask, out, attn);
}
